// GPT_Attention_72584947303141
// MI455X (gfx1250) — matrix operand order repaired and checked by reading, not compiled, not run on hardware
//
#include <hip/hip_runtime.h>

// ---------------------------------------------------------------------------
// GPT attention block for MI455X (gfx1250), wave32 + WMMA f16->f32.
// B=2, S=2048, D=1024, H=16, DH=64.
//
// d_out layout (fp32, concatenated): out[B,S,D] | attn_prob[B,H,S,S] | attn[B,H,S,S]
// d_ws  layout (f16): Qh[B,H,S,DH] | Kh[B,H,S,DH] | Vt[B,H,DH,S] | Ctx[B,S,D]  (32 MB)
//
// Design: memory-bound on the 1.07 GB attn/attn_prob outputs; all matmuls use
// v_wmma_f32_16x16x32_f16; projections use 16x64 per-wave tiles (4 independent
// accumulators per k-step) to keep the matrix pipe saturated.
// ---------------------------------------------------------------------------

typedef _Float16 half_t;
typedef __attribute__((ext_vector_type(16))) _Float16 v16h;
typedef __attribute__((ext_vector_type(8)))  float    v8f;

#define B_  2
#define S_  2048
#define D_  1024
#define H_  16
#define DH_ 64
#define NEGBIG (-1e9f)

// ---- WMMA fragment loaders -------------------------------------------------
// A (16x32 f16, MxK): lanes 0-15 row M=lane, K 0..7 (v0..3) and 16..23 (v4..7);
//                     lanes 16-31 same rows, K 8..15 and 24..31.
// B (32x16 f16, KxN): lane n = column N; lanes 0-15 hold K 0..7 and 16..23,
//                     lanes 16-31 hold K 8..15 and 24..31 (same order as A).

__device__ __forceinline__ v16h frag_a_f32(const float* p0, int ld, int lane) {
  const float* p = p0 + (lane & 15) * ld + ((lane & 16) ? 8 : 0);
  v16h a;
#pragma unroll
  for (int i = 0; i < 8; ++i) { a[i] = (_Float16)p[i]; a[i + 8] = (_Float16)p[i + 16]; }
  return a;
}

__device__ __forceinline__ v16h frag_a_f16(const half_t* p0, int ld, int lane) {
  const half_t* p = p0 + (lane & 15) * ld + ((lane & 16) ? 8 : 0);
  v16h a;
#pragma unroll
  for (int i = 0; i < 8; ++i) { a[i] = p[i]; a[i + 8] = p[i + 16]; }
  return a;
}

// Source is "B transposed": row n holds the K-contiguous column n of B.
__device__ __forceinline__ v16h frag_bt_f32(const float* p0, int ld, int lane) {
  const float* p = p0 + (lane & 15) * ld + ((lane & 16) ? 8 : 0);
  v16h b;
#pragma unroll
  for (int i = 0; i < 8; ++i) { b[i] = (_Float16)p[i]; b[i + 8] = (_Float16)p[i + 16]; }
  return b;
}

__device__ __forceinline__ v16h frag_bt_f16(const half_t* p0, int ld, int lane) {
  const half_t* p = p0 + (lane & 15) * ld + ((lane & 16) ? 8 : 0);
  v16h b;
#pragma unroll
  for (int i = 0; i < 8; ++i) { b[i] = p[i]; b[i + 8] = p[i + 16]; }
  return b;
}

#define WMMA_F32_F16(a, b, c) \
  __builtin_amdgcn_wmma_f32_16x16x32_f16(false, (a), false, (b), (short)0, (c), false, false)

// ---------------------------------------------------------------------------
// Kernel 1: Y = X @ W^T + bias for {q,k,v}; f16 outputs, head-major.
// One wave per 16x64 tile: per k-step, 1 shared A-fragment + 4 independent
// WMMAs (hides XDL latency, quarters A traffic). 32 k-steps over K=1024.
// ---------------------------------------------------------------------------
__global__ __launch_bounds__(256) void qkv_proj_kernel(
    const float* __restrict__ q, const float* __restrict__ k, const float* __restrict__ v,
    const float* __restrict__ Wq, const float* __restrict__ bq,
    const float* __restrict__ Wk, const float* __restrict__ bk,
    const float* __restrict__ Wv, const float* __restrict__ bv,
    half_t* __restrict__ wsQ, half_t* __restrict__ wsK, half_t* __restrict__ wsVt) {
  const int lane = threadIdx.x & 31;
  const int wave = threadIdx.x >> 5;
  const int t    = blockIdx.x * 8 + wave;        // 3 * 256 * 16 = 12288 wave-tiles
  const int mat  = t / 4096;                     // 0=Q 1=K 2=V
  const int rem  = t % 4096;
  const int mt   = rem / (D_ / 64);              // 256 row tiles (B*S/16)
  const int nt4  = rem % (D_ / 64);              // 16 tiles of 64 columns

  const float *X, *W, *bias;
  if (mat == 0)      { X = q; W = Wq; bias = bq; }
  else if (mat == 1) { X = k; W = Wk; bias = bk; }
  else               { X = v; W = Wv; bias = bv; }

  const float* abase = X + (size_t)(mt * 16) * D_;
  const float* bbase = W + (size_t)(nt4 * 64) * D_;  // W row n == column n of W^T

  v8f acc0 = {}, acc1 = {}, acc2 = {}, acc3 = {};
  for (int k0 = 0; k0 < D_; k0 += 32) {
    const v16h af = frag_a_f32(abase + k0, D_, lane);
    const v16h b0 = frag_bt_f32(bbase + k0, D_, lane);
    const v16h b1 = frag_bt_f32(bbase + (size_t)16 * D_ + k0, D_, lane);
    const v16h b2 = frag_bt_f32(bbase + (size_t)32 * D_ + k0, D_, lane);
    const v16h b3 = frag_bt_f32(bbase + (size_t)48 * D_ + k0, D_, lane);
    acc0 = WMMA_F32_F16(af, b0, acc0);
    acc1 = WMMA_F32_F16(af, b1, acc1);
    acc2 = WMMA_F32_F16(af, b2, acc2);
    acc3 = WMMA_F32_F16(af, b3, acc3);
  }

  const float scale = (mat == 0) ? 0.125f : 1.0f;  // Q pre-scaled by 1/sqrt(DH)
  v8f accs[4] = {acc0, acc1, acc2, acc3};
#pragma unroll
  for (int j = 0; j < 4; ++j) {
    const int   n    = nt4 * 64 + j * 16 + (lane & 15);
    const float bval = bias[n];
    const int   h    = n / DH_;
    const int   dh   = n % DH_;
#pragma unroll
    for (int r = 0; r < 8; ++r) {
      const int m  = mt * 16 + r + ((lane & 16) ? 8 : 0);
      const int bb = m / S_;
      const int ss = m % S_;
      const float val = (accs[j][r] + bval) * scale;
      if (mat == 2) {
        wsVt[(((size_t)bb * H_ + h) * DH_ + dh) * S_ + ss] = (half_t)val;  // V transposed
      } else {
        half_t* dst = (mat == 0) ? wsQ : wsK;
        dst[(((size_t)bb * H_ + h) * S_ + ss) * DH_ + dh] = (half_t)val;
      }
    }
  }
}

// ---------------------------------------------------------------------------
// Kernel 2: per (b,h,16-row q-block): scores -> attn, softmax -> attn_prob,
// P @ V -> f16 ctx. 8 waves / workgroup.
// ---------------------------------------------------------------------------
__global__ __launch_bounds__(256) void attn_kernel(
    const half_t* __restrict__ wsQ, const half_t* __restrict__ wsK,
    const half_t* __restrict__ wsVt, const int* __restrict__ mask,
    half_t* __restrict__ wsCtx, float* __restrict__ attn_out,
    float* __restrict__ prob_out) {
  const int lane = threadIdx.x & 31;
  const int wave = threadIdx.x >> 5;
  const int qt   = blockIdx.x % (S_ / 16);
  const int h    = (blockIdx.x / (S_ / 16)) % H_;
  const int b    = blockIdx.x / ((S_ / 16) * H_);
  const size_t bh = (size_t)b * H_ + h;

  const half_t* Qb  = wsQ + (bh * S_ + (size_t)qt * 16) * DH_;
  const half_t* Kb  = wsK + bh * S_ * DH_;
  const half_t* Vtb = wsVt + bh * DH_ * S_;
  float* arow0 = attn_out + (bh * S_ + (size_t)qt * 16) * S_;
  float* prow0 = prob_out + (bh * S_ + (size_t)qt * 16) * S_;

  // ---- phase 1: scores (Q pre-scaled), + additive padding mask ------------
  // Q A-fragments are loop-invariant for the whole workgroup: hoist them.
  const v16h aq0 = frag_a_f16(Qb + 0, DH_, lane);
  const v16h aq1 = frag_a_f16(Qb + 32, DH_, lane);

  // Each wave owns 16 contiguous column tiles, processed 2 at a time
  // -> 4 WMMAs on 2 independent accumulators per iteration.
  for (int ct = wave * 16; ct < wave * 16 + 16; ct += 2) {
    __builtin_prefetch(Kb + (size_t)((ct + 2) & 127) * 16 * DH_, 0, 3);
    v8f acc0 = {}, acc1 = {};
    const half_t* kb0 = Kb + (size_t)(ct * 16) * DH_;
    const half_t* kb1 = kb0 + (size_t)16 * DH_;
    const v16h b00 = frag_bt_f16(kb0 + 0, DH_, lane);
    const v16h b01 = frag_bt_f16(kb0 + 32, DH_, lane);
    const v16h b10 = frag_bt_f16(kb1 + 0, DH_, lane);
    const v16h b11 = frag_bt_f16(kb1 + 32, DH_, lane);
    acc0 = WMMA_F32_F16(aq0, b00, acc0);
    acc1 = WMMA_F32_F16(aq0, b10, acc1);
    acc0 = WMMA_F32_F16(aq1, b01, acc0);
    acc1 = WMMA_F32_F16(aq1, b11, acc1);

#pragma unroll
    for (int j = 0; j < 2; ++j) {
      const v8f&  acc = j ? acc1 : acc0;
      const int   col = (ct + j) * 16 + (lane & 15);
      const float add = (1.0f - (float)mask[b * S_ + col]) * NEGBIG;
#pragma unroll
      for (int r = 0; r < 8; ++r) {
        const int row = r + ((lane & 16) ? 8 : 0);
        arow0[(size_t)row * S_ + col] = acc[r] + add;
      }
    }
  }
  __threadfence();
  __syncthreads();

  // ---- phase 2: row softmax (wave handles 2 rows; shfl_xor reductions) ----
#pragma unroll
  for (int rr = 0; rr < 2; ++rr) {
    const int row = wave * 2 + rr;
    const float* sr = arow0 + (size_t)row * S_;
    float mx = -3.0e38f;
    for (int c = lane; c < S_; c += 32) mx = fmaxf(mx, sr[c]);
#pragma unroll
    for (int off = 16; off > 0; off >>= 1) mx = fmaxf(mx, __shfl_xor(mx, off, 32));
    float sum = 0.0f;
    for (int c = lane; c < S_; c += 32) sum += __expf(sr[c] - mx);
#pragma unroll
    for (int off = 16; off > 0; off >>= 1) sum += __shfl_xor(sum, off, 32);
    const float inv = 1.0f / sum;
    float* dr = prow0 + (size_t)row * S_;
    for (int c = lane; c < S_; c += 32) dr[c] = __expf(sr[c] - mx) * inv;
  }
  __threadfence();
  __syncthreads();

  // ---- phase 3: ctx = P @ V  (waves 0..3, one 16x16 N-tile each; dual
  //               accumulators over even/odd k-steps for WMMA ILP) ----------
  if (wave < 4) {
    const int nt = wave;                       // DH = 4 * 16
    const half_t* vb = Vtb + (size_t)(nt * 16) * S_;
    v8f accE = {}, accO = {};
    for (int k0 = 0; k0 < S_; k0 += 64) {
      const int ko = ((lane & 16) ? 8 : 0);
      const float* prE = prow0 + (size_t)(lane & 15) * S_ + k0 + ko;
      const float* prO = prE + 32;
      v16h aE, aO;
#pragma unroll
      for (int i = 0; i < 8; ++i) {
        aE[i] = (_Float16)prE[i]; aE[i + 8] = (_Float16)prE[i + 16];
        aO[i] = (_Float16)prO[i]; aO[i + 8] = (_Float16)prO[i + 16];
      }
      const v16h bE = frag_bt_f16(vb + k0, S_, lane);       // Vt row dh = col dh of V
      const v16h bO = frag_bt_f16(vb + k0 + 32, S_, lane);
      accE = WMMA_F32_F16(aE, bE, accE);
      accO = WMMA_F32_F16(aO, bO, accO);
    }
    const int dh = nt * 16 + (lane & 15);
#pragma unroll
    for (int r = 0; r < 8; ++r) {
      const int row = r + ((lane & 16) ? 8 : 0);
      const int s   = qt * 16 + row;
      wsCtx[((size_t)b * S_ + s) * D_ + h * DH_ + dh] = (half_t)(accE[r] + accO[r]);
    }
  }
}

// ---------------------------------------------------------------------------
// Kernel 3: out = ctx(f16) @ Wfc^T + bfc  (fp32 out), 16x64 tile per wave.
// ---------------------------------------------------------------------------
__global__ __launch_bounds__(256) void final_proj_kernel(
    const half_t* __restrict__ wsCtx, const float* __restrict__ Wfc,
    const float* __restrict__ bfc, float* __restrict__ out) {
  const int lane = threadIdx.x & 31;
  const int wave = threadIdx.x >> 5;
  const int t   = blockIdx.x * 8 + wave;       // 256 * 16 = 4096 wave-tiles
  const int mt  = t / (D_ / 64);
  const int nt4 = t % (D_ / 64);

  const half_t* abase = wsCtx + (size_t)(mt * 16) * D_;
  const float*  bbase = Wfc + (size_t)(nt4 * 64) * D_;

  v8f acc0 = {}, acc1 = {}, acc2 = {}, acc3 = {};
  for (int k0 = 0; k0 < D_; k0 += 32) {
    const v16h af = frag_a_f16(abase + k0, D_, lane);
    const v16h b0 = frag_bt_f32(bbase + k0, D_, lane);
    const v16h b1 = frag_bt_f32(bbase + (size_t)16 * D_ + k0, D_, lane);
    const v16h b2 = frag_bt_f32(bbase + (size_t)32 * D_ + k0, D_, lane);
    const v16h b3 = frag_bt_f32(bbase + (size_t)48 * D_ + k0, D_, lane);
    acc0 = WMMA_F32_F16(af, b0, acc0);
    acc1 = WMMA_F32_F16(af, b1, acc1);
    acc2 = WMMA_F32_F16(af, b2, acc2);
    acc3 = WMMA_F32_F16(af, b3, acc3);
  }

  v8f accs[4] = {acc0, acc1, acc2, acc3};
#pragma unroll
  for (int j = 0; j < 4; ++j) {
    const int   n    = nt4 * 64 + j * 16 + (lane & 15);
    const float bias = bfc[n];
#pragma unroll
    for (int r = 0; r < 8; ++r) {
      const int m = mt * 16 + r + ((lane & 16) ? 8 : 0);
      out[(size_t)m * D_ + n] = accs[j][r] + bias;
    }
  }
}

// ---------------------------------------------------------------------------
extern "C" void kernel_launch(void* const* d_in, const int* in_sizes, int n_in,
                              void* d_out, int out_size, void* d_ws, size_t ws_size,
                              hipStream_t stream) {
  const float* q    = (const float*)d_in[0];
  const float* k    = (const float*)d_in[1];
  const float* v    = (const float*)d_in[2];
  const int*   mask = (const int*)d_in[3];
  const float* Wq   = (const float*)d_in[4];
  const float* bq   = (const float*)d_in[5];
  const float* Wk   = (const float*)d_in[6];
  const float* bk   = (const float*)d_in[7];
  const float* Wv   = (const float*)d_in[8];
  const float* bv   = (const float*)d_in[9];
  const float* Wfc  = (const float*)d_in[10];
  const float* bfc  = (const float*)d_in[11];

  float* out  = (float*)d_out;
  float* prob = out + (size_t)B_ * S_ * D_;
  float* attn = prob + (size_t)B_ * H_ * S_ * S_;

  half_t* wsQ   = (half_t*)d_ws;
  half_t* wsK   = wsQ + (size_t)B_ * H_ * S_ * DH_;
  half_t* wsVt  = wsK + (size_t)B_ * H_ * S_ * DH_;
  half_t* wsCtx = wsVt + (size_t)B_ * H_ * S_ * DH_;

  // 3 * 256 * 16 wave-tiles (16x64 each), 8 waves per block
  qkv_proj_kernel<<<1536, 256, 0, stream>>>(q, k, v, Wq, bq, Wk, bk, Wv, bv,
                                            wsQ, wsK, wsVt);
  // one workgroup per (b, h, 16-row query block)
  attn_kernel<<<B_ * H_ * (S_ / 16), 256, 0, stream>>>(wsQ, wsK, wsVt, mask,
                                                       wsCtx, attn, prob);
  // 256 * 16 wave-tiles (16x64 each), 8 waves per block
  final_proj_kernel<<<512, 256, 0, stream>>>(wsCtx, Wfc, bfc, out);
}
